// GemmaForCausalLM_35055523070018
// MI455X (gfx1250) — matrix operand order repaired and checked by reading, not compiled, not run on hardware
//
#include <hip/hip_runtime.h>
#include <hip/hip_bf16.h>
#include <math.h>

// ---------------- model constants ----------------
#define T_   1024
#define H_   2048
#define V_   32000
#define NH_  8
#define NKV_ 4
#define HD_  256
#define I_   8192
#define L_   2
#define EPS_ 1e-6f
#define SCALE_ 0.0625f          // 256^-0.5
#define NEGBIG_ -1.0e30f

typedef long long i64;
typedef __bf16 bf16_t;
typedef bf16_t v16bf __attribute__((ext_vector_type(16)));
typedef bf16_t v8bf  __attribute__((ext_vector_type(8)));
typedef float  v8f   __attribute__((ext_vector_type(8)));
typedef float  v16f  __attribute__((ext_vector_type(16)));

// native f32 -> bf16 (RNE); lowers to v_cvt_*bf16* on gfx1250
__device__ __forceinline__ bf16_t f2bf(float f) { return (bf16_t)f; }

// =====================================================================
// WMMA GEMM: C[M,N](f32) = A[M,K](bf16,row-major) * B
//   BMODE 0: B = f32  [N,K] row-major (k-contiguous)  -> in-register bf16 cvt
//   BMODE 1: B = bf16 [N,K] row-major (k-contiguous)
//   BMODE 2: B = bf16 [K,N] row-major (n-contiguous)
// Block = 128 threads = 4 waves (2x2); block tile 64(M) x 128(N);
// wave tile 32x64 = 2 A frags x 4 B frags = 8 v_wmma per K=32 step.
// z-batched (attention heads): A += aOffZ*z, B += bOffZ*(z>>bShift), C += cOffZ*z
// =====================================================================
template<int BMODE>
__global__ __launch_bounds__(128)
void gemm_wmma(const bf16_t* __restrict__ A, int lda, i64 aOffZ,
               const void* __restrict__ Bv, int ldb, i64 bOffZ, int bShift,
               float* __restrict__ C, int ldc, i64 cOffZ, int K)
{
  const int z    = blockIdx.z;
  const int lane = threadIdx.x & 31;
  const int wave = threadIdx.x >> 5;
  const int l15  = lane & 15;
  const int kh   = lane >> 4;                       // lane half selects K sub-range
  const int m0   = blockIdx.y * 64  + (wave >> 1) * 32;
  const int nw   = blockIdx.x * 128 + (wave & 1) * 64;

  const bf16_t* Ab = A + aOffZ * (i64)z;
  C += cOffZ * (i64)z;

  const float*  Bf = nullptr;
  const bf16_t* Bh = nullptr;
  if constexpr (BMODE == 0) Bf = (const float*) Bv + bOffZ * (i64)(z >> bShift);
  else                      Bh = (const bf16_t*)Bv + bOffZ * (i64)(z >> bShift);

  v8f acc[2][4] = {};

  const bf16_t* arow0 = Ab + (i64)(m0 + l15)      * lda;
  const bf16_t* arow1 = Ab + (i64)(m0 + 16 + l15) * lda;

  for (int kk = 0; kk < K; kk += 32) {
    // ---- A fragments (K = kh*16 + e, matching B) ----
    v16bf a[2];
    {
      a[0] = *(const v16bf*)(arow0 + kk + kh * 16);
      a[1] = *(const v16bf*)(arow1 + kk + kh * 16);
    }

    // ---- B fragments (col = n0 + l15, K = kh*16 + e) ----
    v16bf b[4];
    #pragma unroll
    for (int f = 0; f < 4; ++f) {
      const int n0 = nw + f * 16;
      if constexpr (BMODE == 0) {
        const float* p = Bf + (i64)(n0 + l15) * ldb + kk + kh * 16;
        __builtin_prefetch(p + 32, 0, 1);           // global_prefetch_b8: next K panel
        v16f t = *(const v16f*)p;                   // 4x global_load_b128
        b[f] = __builtin_convertvector(t, v16bf);   // packed f32->bf16 cvt
      } else if constexpr (BMODE == 1) {
        b[f] = *(const v16bf*)(Bh + (i64)(n0 + l15) * ldb + kk + kh * 16);
      } else {
        const bf16_t* p = Bh + (i64)(kk + kh * 16) * ldb + n0 + l15;
        #pragma unroll
        for (int i = 0; i < 16; ++i) b[f][i] = p[(i64)i * ldb];
      }
    }

    #pragma unroll
    for (int mi = 0; mi < 2; ++mi)
      #pragma unroll
      for (int f = 0; f < 4; ++f)
        acc[mi][f] = __builtin_amdgcn_wmma_f32_16x16x32_bf16(
            false, a[mi], false, b[f], (short)0, acc[mi][f], false, false);
  }

  // ---- store: D element e -> row m_base + kh*8 + e, col n + l15 ----
  #pragma unroll
  for (int mi = 0; mi < 2; ++mi) {
    float* c0 = C + (i64)(m0 + mi * 16 + kh * 8) * ldc + nw + l15;
    #pragma unroll
    for (int f = 0; f < 4; ++f)
      #pragma unroll
      for (int e = 0; e < 8; ++e)
        c0[(i64)e * ldc + f * 16] = acc[mi][f][e];
  }
}

// ---------------- elementwise / reduction kernels ----------------
__global__ __launch_bounds__(256)
void embed_kernel(const int* __restrict__ tokens, const float* __restrict__ embed,
                  float* __restrict__ x)
{
  int t = blockIdx.x;
  const float* e = embed + (i64)tokens[t] * H_;
  float* xr = x + (i64)t * H_;
  for (int h = threadIdx.x; h < H_; h += 256) xr[h] = e[h] * 45.254834f; // sqrt(2048)
}

__global__ __launch_bounds__(256)
void rmsnorm_bf16_kernel(const float* __restrict__ x, const float* __restrict__ w,
                         bf16_t* __restrict__ out)
{
  __shared__ float red[256];
  int t = blockIdx.x;
  const float* xr = x + (i64)t * H_;
  float ss = 0.f;
  for (int h = threadIdx.x; h < H_; h += 256) { float v = xr[h]; ss += v * v; }
  red[threadIdx.x] = ss; __syncthreads();
  for (int s = 128; s > 0; s >>= 1) {
    if (threadIdx.x < s) red[threadIdx.x] += red[threadIdx.x + s];
    __syncthreads();
  }
  float inv = rsqrtf(red[0] * (1.0f / H_) + EPS_);
  bf16_t* orow = out + (i64)t * H_;
  for (int h = threadIdx.x; h < H_; h += 256)
    orow[h] = f2bf(xr[h] * inv * (1.f + w[h]));
}

__global__ __launch_bounds__(256)
void rmsnorm_add_kernel(float* __restrict__ x, const float* __restrict__ y,
                        const float* __restrict__ w)
{
  __shared__ float red[256];
  int t = blockIdx.x;
  const float* yr = y + (i64)t * H_;
  float ss = 0.f;
  for (int h = threadIdx.x; h < H_; h += 256) { float v = yr[h]; ss += v * v; }
  red[threadIdx.x] = ss; __syncthreads();
  for (int s = 128; s > 0; s >>= 1) {
    if (threadIdx.x < s) red[threadIdx.x] += red[threadIdx.x + s];
    __syncthreads();
  }
  float inv = rsqrtf(red[0] * (1.0f / H_) + EPS_);
  float* xr = x + (i64)t * H_;
  for (int h = threadIdx.x; h < H_; h += 256)
    xr[h] += yr[h] * inv * (1.f + w[h]);
}

// qkv[T,4096] -> rope(Q)[T,2048]bf16, rope(K)[T,1024]bf16, V[T,1024]bf16
__global__ __launch_bounds__(256)
void rope_split_kernel(const float* __restrict__ qkv, bf16_t* __restrict__ Qb,
                       bf16_t* __restrict__ Kb, bf16_t* __restrict__ Vb)
{
  int t = blockIdx.x;
  const float* row = qkv + (i64)t * 4096;
  for (int idx = threadIdx.x; idx < 1024; idx += 256) {
    int h = idx >> 7, j = idx & 127;
    float freq = __expf(-0.07195578f * (float)j);   // ln(1e4)/128
    float ang = (float)t * freq;
    float s, c; __sincosf(ang, &s, &c);
    float q1 = row[h * 256 + j], q2 = row[h * 256 + 128 + j];
    Qb[(i64)t * 2048 + h * 256 + j]       = f2bf(q1 * c - q2 * s);
    Qb[(i64)t * 2048 + h * 256 + 128 + j] = f2bf(q1 * s + q2 * c);
    if (h < NKV_) {
      float k1 = row[2048 + h * 256 + j], k2 = row[2048 + h * 256 + 128 + j];
      Kb[(i64)t * 1024 + h * 256 + j]       = f2bf(k1 * c - k2 * s);
      Kb[(i64)t * 1024 + h * 256 + 128 + j] = f2bf(k1 * s + k2 * c);
    }
    Vb[(i64)t * 1024 + idx] = f2bf(row[3072 + idx]);
  }
}

// S[NH,T,T] f32 (raw q.k) -> P[NH,T,T] bf16 with scale, softcap, mask, softmax
__global__ __launch_bounds__(256)
void softmax_kernel(const float* __restrict__ S, bf16_t* __restrict__ P, int win)
{
  __shared__ float red[256];
  int i = blockIdx.x, h = blockIdx.y, tid = threadIdx.x;
  const float* srow = S + ((i64)h * T_ + i) * T_;
  bf16_t*      prow = P + ((i64)h * T_ + i) * T_;
  float v[4];
  float mx = -3.0e38f;
  #pragma unroll
  for (int r = 0; r < 4; ++r) {
    int j = r * 256 + tid;
    bool ok = (j <= i) && ((i - j) < win);
    float sv = srow[j] * SCALE_;
    sv = tanhf(sv * 0.02f) * 50.f;                  // tanh(s/50)*50
    v[r] = ok ? sv : NEGBIG_;
    mx = fmaxf(mx, v[r]);
  }
  red[tid] = mx; __syncthreads();
  for (int s = 128; s > 0; s >>= 1) {
    if (tid < s) red[tid] = fmaxf(red[tid], red[tid + s]);
    __syncthreads();
  }
  float m = red[0]; __syncthreads();
  float sum = 0.f;
  #pragma unroll
  for (int r = 0; r < 4; ++r) { v[r] = __expf(v[r] - m); sum += v[r]; }
  red[tid] = sum; __syncthreads();
  for (int s = 128; s > 0; s >>= 1) {
    if (tid < s) red[tid] += red[tid + s];
    __syncthreads();
  }
  float inv = 1.f / red[0];
  #pragma unroll
  for (int r = 0; r < 4; ++r) prow[r * 256 + tid] = f2bf(v[r] * inv);
}

__global__ __launch_bounds__(256)
void gelu_mul_kernel(const float* __restrict__ g, const float* __restrict__ u,
                     bf16_t* __restrict__ out, i64 n)
{
  for (i64 i = (i64)blockIdx.x * 256 + threadIdx.x; i < n; i += (i64)gridDim.x * 256) {
    float x = g[i];
    float t = tanhf(0.79788456f * (x + 0.044715f * x * x * x));
    out[i] = f2bf(0.5f * x * (1.f + t) * u[i]);
  }
}

__global__ __launch_bounds__(256)
void cvt_bf16_kernel(const float* __restrict__ in, bf16_t* __restrict__ out, i64 n)
{
  for (i64 i = (i64)blockIdx.x * 256 + threadIdx.x; i < n; i += (i64)gridDim.x * 256)
    out[i] = f2bf(in[i]);
}

__global__ __launch_bounds__(256)
void cap_kernel(float* __restrict__ x, i64 n)
{
  for (i64 i = (i64)blockIdx.x * 256 + threadIdx.x; i < n; i += (i64)gridDim.x * 256)
    x[i] = tanhf(x[i] * (1.f / 30.f)) * 30.f;
}

// ---------------- launcher ----------------
extern "C" void kernel_launch(void* const* d_in, const int* in_sizes, int n_in,
                              void* d_out, int out_size, void* d_ws, size_t ws_size,
                              hipStream_t stream)
{
  (void)in_sizes; (void)n_in; (void)out_size; (void)ws_size;
  const int*   tokens  = (const int*)  d_in[0];
  const float* embed   = (const float*)d_in[1];
  const float* qkv_w   = (const float*)d_in[2];
  const float* o_w     = (const float*)d_in[3];
  const float* gate_w  = (const float*)d_in[4];
  const float* up_w    = (const float*)d_in[5];
  const float* down_w  = (const float*)d_in[6];
  const float* ln_in   = (const float*)d_in[7];
  const float* ln_pa   = (const float*)d_in[8];
  const float* ln_pre  = (const float*)d_in[9];
  const float* ln_post = (const float*)d_in[10];
  const float* ln_f    = (const float*)d_in[11];
  float* out = (float*)d_out;

  char* ws = (char*)d_ws;
  size_t off = 0;
  auto alloc = [&](size_t bytes) -> void* {
    void* p = ws + off;
    off += (bytes + 255) & ~(size_t)255;
    return p;
  };
  float*  x    = (float*) alloc((size_t)T_ * H_ * 4);
  bf16_t* hb   = (bf16_t*)alloc((size_t)T_ * H_ * 2);
  float*  qkv  = (float*) alloc((size_t)T_ * 4096 * 4);
  bf16_t* Qb   = (bf16_t*)alloc((size_t)T_ * 2048 * 2);
  bf16_t* Kb   = (bf16_t*)alloc((size_t)T_ * 1024 * 2);
  bf16_t* Vb   = (bf16_t*)alloc((size_t)T_ * 1024 * 2);
  float*  S    = (float*) alloc((size_t)NH_ * T_ * T_ * 4);
  bf16_t* P    = (bf16_t*)alloc((size_t)NH_ * T_ * T_ * 2);
  float*  AO   = (float*) alloc((size_t)T_ * H_ * 4);
  bf16_t* AOb  = (bf16_t*)alloc((size_t)T_ * H_ * 2);
  float*  tmpH = (float*) alloc((size_t)T_ * H_ * 4);
  float*  gate = (float*) alloc((size_t)T_ * I_ * 4);
  float*  up   = (float*) alloc((size_t)T_ * I_ * 4);
  bf16_t* gu   = (bf16_t*)alloc((size_t)T_ * I_ * 2);

  dim3 blk(128);
  auto G = [](int N, int M, int Z) { return dim3(N / 128, M / 64, Z); };

  embed_kernel<<<T_, 256, 0, stream>>>(tokens, embed, x);

  for (int l = 0; l < L_; ++l) {
    // --- attention ---
    rmsnorm_bf16_kernel<<<T_, 256, 0, stream>>>(x, ln_in + (i64)l * H_, hb);
    gemm_wmma<0><<<G(4096, T_, 1), blk, 0, stream>>>(
        hb, H_, 0, qkv_w + (i64)l * 4096 * H_, H_, 0, 0, qkv, 4096, 0, H_);
    rope_split_kernel<<<T_, 256, 0, stream>>>(qkv, Qb, Kb, Vb);
    // scores: per head z, A=Q(head z), B=K(head z/2), K=HD
    gemm_wmma<1><<<G(T_, T_, NH_), blk, 0, stream>>>(
        Qb, 2048, 256, Kb, 1024, 256, 1, S, T_, (i64)T_ * T_, HD_);
    softmax_kernel<<<dim3(T_, NH_), 256, 0, stream>>>(S, P, l == 0 ? 512 : (1 << 30));
    // out = P @ V: B is [K,N] bf16
    gemm_wmma<2><<<G(HD_, T_, NH_), blk, 0, stream>>>(
        P, T_, (i64)T_ * T_, Vb, 1024, 256, 1, AO, 2048, 256, T_);
    cvt_bf16_kernel<<<2048, 256, 0, stream>>>(AO, AOb, (i64)T_ * H_);
    gemm_wmma<0><<<G(H_, T_, 1), blk, 0, stream>>>(
        AOb, 2048, 0, o_w + (i64)l * H_ * 2048, 2048, 0, 0, tmpH, H_, 0, 2048);
    rmsnorm_add_kernel<<<T_, 256, 0, stream>>>(x, tmpH, ln_pa + (i64)l * H_);

    // --- MLP ---
    rmsnorm_bf16_kernel<<<T_, 256, 0, stream>>>(x, ln_pre + (i64)l * H_, hb);
    gemm_wmma<0><<<G(I_, T_, 1), blk, 0, stream>>>(
        hb, H_, 0, gate_w + (i64)l * I_ * H_, H_, 0, 0, gate, I_, 0, H_);
    gemm_wmma<0><<<G(I_, T_, 1), blk, 0, stream>>>(
        hb, H_, 0, up_w + (i64)l * I_ * H_, H_, 0, 0, up, I_, 0, H_);
    gelu_mul_kernel<<<4096, 256, 0, stream>>>(gate, up, gu, (i64)T_ * I_);
    gemm_wmma<0><<<G(H_, T_, 1), blk, 0, stream>>>(
        gu, I_, 0, down_w + (i64)l * H_ * I_, I_, 0, 0, tmpH, H_, 0, I_);
    rmsnorm_add_kernel<<<T_, 256, 0, stream>>>(x, tmpH, ln_post + (i64)l * H_);
  }

  // --- final norm + logits + cap ---
  rmsnorm_bf16_kernel<<<T_, 256, 0, stream>>>(x, ln_f, hb);
  gemm_wmma<0><<<G(V_, T_, 1), blk, 0, stream>>>(
      hb, H_, 0, embed, H_, 0, 0, out, V_, 0, H_);
  cap_kernel<<<4096, 256, 0, stream>>>(out, (i64)T_ * V_);
}
